// AttentionalLatentTrajectoryGenerator_927712936386
// MI455X (gfx1250) — compile-verified, matrix operand order checked by reading, not run on hardware
//
#include <hip/hip_runtime.h>
#include <hip/hip_bf16.h>

// ---------------------------------------------------------------------------
// AttentionalLatentTrajectoryGenerator for MI455X (gfx1250).
//
// Key algebraic simplification: seq_len==1 self-attention => softmax over one
// key == 1.0, so q/k projections (Wq,bq,Wk,bk) are dead and
//   attn(x) = x @ (Wv@Wo) + (bv@Wo + bo)
// We fold Wvo/bvo once per launch on-device.
//
// All GEMMs run on v_wmma_f32_16x16x32_bf16 (wave32, fp32 accumulate).
// Weights are repacked to [N,K] bf16 so A (activations, row-major [M,K]) and
// B (transposed weights) tiles both load as two contiguous b128 per lane,
// matching the CDNA5 16-bit A/B operand layout:
//   lanes 0-15  : row M=lane,    K in {0..7} U {16..23}
//   lanes 16-31 : row M=lane-16, K in {8..15} U {24..31}
// ---------------------------------------------------------------------------

#define HID 1024
#define LAT 256
#define BATCH 64

typedef __bf16 bf16;
typedef __attribute__((ext_vector_type(16))) __bf16 v16bf;
typedef __attribute__((ext_vector_type(8)))  __bf16 v8bf;
typedef __attribute__((ext_vector_type(8)))  float  v8f;

__device__ __forceinline__ v16bf load_tile(const bf16* __restrict__ base, int ld,
                                           int r0, int k0) {
  const int lane = threadIdx.x & 31;
  const int lr   = lane & 15;
  const int half = lane >> 4;
  const bf16* p = base + (size_t)(r0 + lr) * ld + k0 + half * 8;
  v8bf lo = *(const v8bf*)(p);        // K = k0 + half*8 + {0..7}
  v8bf hi = *(const v8bf*)(p + 16);   // K = k0 + 16 + half*8 + {0..7}
  return __builtin_shufflevector(lo, hi, 0,1,2,3,4,5,6,7,8,9,10,11,12,13,14,15);
}

__device__ __forceinline__ v8f wmma_bf16(v16bf a, v16bf b, v8f c) {
  return __builtin_amdgcn_wmma_f32_16x16x32_bf16(false, a, false, b,
                                                 (short)0, c, false, false);
}

__device__ __forceinline__ float sigmoidf_(float x) {
  return 1.0f / (1.0f + __expf(-x));
}

// ---------------------------------------------------------------------------
// Generic tiled GEMM: C[M,N] = A[M,K] @ Bt[N,K]^T + bias, optional ReLU,
// optional bf16 and/or f32 stores. One wave per 16x16 tile.
// grid = (N/16, M/16), block = 32.
// ---------------------------------------------------------------------------
template <int RELU, int SBF, int SF32>
__global__ void k_gemm(const bf16* __restrict__ A, const bf16* __restrict__ Bt,
                       const float* __restrict__ bias, int K,
                       bf16* __restrict__ obf, int ldbf,
                       float* __restrict__ of, int ldf) {
  const int col0 = blockIdx.x * 16;
  const int row0 = blockIdx.y * 16;
  const int lane = threadIdx.x & 31;
  const int ln = lane & 15, half = lane >> 4;

  v8f acc = {};
#pragma unroll 4
  for (int k0 = 0; k0 < K; k0 += 32) {
    v16bf a = load_tile(A, K, row0, k0);
    v16bf b = load_tile(Bt, K, col0, k0);
    acc = wmma_bf16(a, b, acc);
  }

  const int col = col0 + ln;
  const float bb = bias[col];
#pragma unroll
  for (int r = 0; r < 8; ++r) {
    const int row = row0 + half * 8 + r;
    float v = acc[r] + bb;
    if (RELU) v = fmaxf(v, 0.0f);
    if (SBF)  obf[(size_t)row * ldbf + col] = (bf16)v;
    if (SF32) of[(size_t)row * ldf + col] = v;
  }
}

// ---------------------------------------------------------------------------
// Fused GRU layer: each wave owns one 16x16 tile of h_new and accumulates the
// six gate GEMM tiles (x@Wih and h@Whh for r/z/n at column offsets
// +0/+HID/+2*HID) in registers, then applies the gate math.
// grid = (HID/16, BATCH/16), block = 32.
// ---------------------------------------------------------------------------
__global__ void k_gru(const bf16* __restrict__ x, const bf16* __restrict__ hbf,
                      const float* __restrict__ hf,
                      const bf16* __restrict__ WihT, const bf16* __restrict__ WhhT,
                      const float* __restrict__ bih, const float* __restrict__ bhh,
                      float* __restrict__ hof, bf16* __restrict__ hobf) {
  const int col0 = blockIdx.x * 16;
  const int row0 = blockIdx.y * 16;
  const int lane = threadIdx.x & 31;
  const int ln = lane & 15, half = lane >> 4;

  v8f gi_r = {}, gi_z = {}, gi_n = {};
  v8f gh_r = {}, gh_z = {}, gh_n = {};

#pragma unroll 2
  for (int k0 = 0; k0 < HID; k0 += 32) {
    v16bf ax = load_tile(x,   HID, row0, k0);
    v16bf ah = load_tile(hbf, HID, row0, k0);
    v16bf br = load_tile(WihT, HID, col0,           k0);
    v16bf bz = load_tile(WihT, HID, col0 + HID,     k0);
    v16bf bn = load_tile(WihT, HID, col0 + 2 * HID, k0);
    gi_r = wmma_bf16(ax, br, gi_r);
    gi_z = wmma_bf16(ax, bz, gi_z);
    gi_n = wmma_bf16(ax, bn, gi_n);
    v16bf cr = load_tile(WhhT, HID, col0,           k0);
    v16bf cz = load_tile(WhhT, HID, col0 + HID,     k0);
    v16bf cn = load_tile(WhhT, HID, col0 + 2 * HID, k0);
    gh_r = wmma_bf16(ah, cr, gh_r);
    gh_z = wmma_bf16(ah, cz, gh_z);
    gh_n = wmma_bf16(ah, cn, gh_n);
  }

  const int col = col0 + ln;
  const float bir = bih[col], biz = bih[col + HID], bin = bih[col + 2 * HID];
  const float bhr = bhh[col], bhz = bhh[col + HID], bhn = bhh[col + 2 * HID];

#pragma unroll
  for (int r = 0; r < 8; ++r) {
    const int row = row0 + half * 8 + r;
    const float ir = gi_r[r] + bir, iz = gi_z[r] + biz, in = gi_n[r] + bin;
    const float hr = gh_r[r] + bhr, hz = gh_z[r] + bhz, hn = gh_n[r] + bhn;
    const float rg = sigmoidf_(ir + hr);
    const float zg = sigmoidf_(iz + hz);
    const float ng = tanhf(in + rg * hn);
    const float hold = hf[(size_t)row * HID + col];
    const float hnew = (1.0f - zg) * ng + zg * hold;
    hof[(size_t)row * HID + col]  = hnew;
    hobf[(size_t)row * HID + col] = (bf16)hnew;
  }
}

// ------------------------- prep / utility kernels --------------------------

// Wt[n*K + k] = bf16(W[k*N + n])   (W is [K,N] row-major)
__global__ void k_transpose(const float* __restrict__ W, bf16* __restrict__ Wt,
                            int K, int N) {
  const int i = blockIdx.x * blockDim.x + threadIdx.x;
  if (i >= K * N) return;
  const int n = i / K, k = i - n * K;
  Wt[i] = (bf16)W[(size_t)k * N + n];
}

// WvoT[n*HID + k] = bf16( sum_j Wv[k,j] * Wo[j,n] )
__global__ void k_wvo(const float* __restrict__ Wv, const float* __restrict__ Wo,
                      bf16* __restrict__ WvoT) {
  const int i = blockIdx.x * blockDim.x + threadIdx.x;
  const int n = i & (HID - 1);
  const int k = i >> 10;
  float s = 0.0f;
  for (int j = 0; j < HID; ++j)
    s = fmaf(Wv[(size_t)k * HID + j], Wo[(size_t)j * HID + n], s);
  WvoT[(size_t)n * HID + k] = (bf16)s;
}

// bvo[n] = bo[n] + sum_j bv[j] * Wo[j,n]
__global__ void k_bvo(const float* __restrict__ bv, const float* __restrict__ Wo,
                      const float* __restrict__ bo, float* __restrict__ bvo) {
  const int n = blockIdx.x * blockDim.x + threadIdx.x;
  if (n >= HID) return;
  float s = bo[n];
  for (int j = 0; j < HID; ++j) s = fmaf(bv[j], Wo[(size_t)j * HID + n], s);
  bvo[n] = s;
}

__global__ void k_cast(const float* __restrict__ src, bf16* __restrict__ dst, int n) {
  const int i = blockIdx.x * blockDim.x + threadIdx.x;
  if (i < n) dst[i] = (bf16)src[i];
}

// replicate initial state: h2f = h0f ; h1bf = h2bf = gin_bf
__global__ void k_fanout(const float* __restrict__ h0f, const bf16* __restrict__ ginbf,
                         float* __restrict__ h2f, bf16* __restrict__ h1bf,
                         bf16* __restrict__ h2bf) {
  const int i = blockIdx.x * blockDim.x + threadIdx.x;
  if (i >= BATCH * HID) return;
  h2f[i] = h0f[i];
  const bf16 g = ginbf[i];
  h1bf[i] = g;
  h2bf[i] = g;
}

// ---------------------------------------------------------------------------

extern "C" void kernel_launch(void* const* d_in, const int* in_sizes, int n_in,
                              void* d_out, int out_size, void* d_ws, size_t ws_size,
                              hipStream_t stream) {
  const float* z_start = (const float*)d_in[0];
  // d_in[1] = max_len (device scalar); T recovered from out_size instead.
  const float* w1   = (const float*)d_in[2];
  const float* b1   = (const float*)d_in[3];
  const float* w2   = (const float*)d_in[4];
  const float* b2   = (const float*)d_in[5];
  // d_in[6..9] = Wq,bq,Wk,bk : mathematically dead (softmax over one key).
  const float* Wv   = (const float*)d_in[10];
  const float* bv   = (const float*)d_in[11];
  const float* Wo   = (const float*)d_in[12];
  const float* bo   = (const float*)d_in[13];
  const float* Wih0 = (const float*)d_in[14];
  const float* Whh0 = (const float*)d_in[15];
  const float* bih0 = (const float*)d_in[16];
  const float* bhh0 = (const float*)d_in[17];
  const float* Wih1 = (const float*)d_in[18];
  const float* Whh1 = (const float*)d_in[19];
  const float* bih1 = (const float*)d_in[20];
  const float* bhh1 = (const float*)d_in[21];
  const float* Wh   = (const float*)d_in[22];
  const float* bh   = (const float*)d_in[23];

  float* outp = (float*)d_out;
  const int T = out_size / (BATCH * LAT);

  // ---- workspace carve-out (all offsets 256B aligned for b128 tile loads)
  char* ws = (char*)d_ws;
  size_t off = 0;
  auto alloc = [&](size_t bytes) -> void* {
    void* p = ws + off;
    off = (off + bytes + 255) & ~(size_t)255;
    return p;
  };
  bf16* WvoT  = (bf16*)alloc((size_t)HID * HID * 2);
  bf16* Wih0T = (bf16*)alloc((size_t)3 * HID * HID * 2);
  bf16* Whh0T = (bf16*)alloc((size_t)3 * HID * HID * 2);
  bf16* Wih1T = (bf16*)alloc((size_t)3 * HID * HID * 2);
  bf16* Whh1T = (bf16*)alloc((size_t)3 * HID * HID * 2);
  bf16* WhT   = (bf16*)alloc((size_t)LAT * HID * 2);
  bf16* W1T   = (bf16*)alloc((size_t)HID * LAT * 2);
  bf16* W2T   = (bf16*)alloc((size_t)HID * HID * 2);
  float* bvo  = (float*)alloc(HID * 4);
  bf16* zs_bf = (bf16*)alloc((size_t)BATCH * LAT * 2);
  bf16* gin_bf = (bf16*)alloc((size_t)BATCH * HID * 2);
  bf16* a_bf   = (bf16*)alloc((size_t)BATCH * HID * 2);
  bf16* t1_bf  = (bf16*)alloc((size_t)BATCH * HID * 2);
  bf16* nz_bf  = (bf16*)alloc((size_t)BATCH * LAT * 2);
  float* h1f[2] = {(float*)alloc((size_t)BATCH * HID * 4),
                   (float*)alloc((size_t)BATCH * HID * 4)};
  float* h2f[2] = {(float*)alloc((size_t)BATCH * HID * 4),
                   (float*)alloc((size_t)BATCH * HID * 4)};
  bf16* h1bf[2] = {(bf16*)alloc((size_t)BATCH * HID * 2),
                   (bf16*)alloc((size_t)BATCH * HID * 2)};
  bf16* h2bf[2] = {(bf16*)alloc((size_t)BATCH * HID * 2),
                   (bf16*)alloc((size_t)BATCH * HID * 2)};
  if (off > ws_size) return;  // workspace too small; bail deterministically

  // ---- per-launch weight prep (runs in ~tens of us; weights then L2-resident)
  auto tpose = [&](const float* W, bf16* Wt, int K, int N) {
    const int n = K * N;
    k_transpose<<<(n + 255) / 256, 256, 0, stream>>>(W, Wt, K, N);
  };
  tpose(Wih0, Wih0T, HID, 3 * HID);
  tpose(Whh0, Whh0T, HID, 3 * HID);
  tpose(Wih1, Wih1T, HID, 3 * HID);
  tpose(Whh1, Whh1T, HID, 3 * HID);
  tpose(Wh,   WhT,   HID, LAT);
  tpose(w1,   W1T,   LAT, HID);
  tpose(w2,   W2T,   HID, HID);
  k_wvo<<<(HID * HID) / 256, 256, 0, stream>>>(Wv, Wo, WvoT);
  k_bvo<<<(HID + 255) / 256, 256, 0, stream>>>(bv, Wo, bo, bvo);
  k_cast<<<(BATCH * LAT + 255) / 256, 256, 0, stream>>>(z_start, zs_bf, BATCH * LAT);

  const dim3 gHID(HID / 16, BATCH / 16);  // 64 x 4 tiles
  const dim3 gLAT(LAT / 16, BATCH / 16);  // 16 x 4 tiles

  // ---- init: h0 = relu(z@w1+b1)@w2 + b2 ; h1=h2=gin=h0 (NO outer relu)
  k_gemm<1, 1, 0><<<gHID, 32, 0, stream>>>(zs_bf, W1T, b1, LAT, t1_bf, HID, nullptr, 0);
  k_gemm<0, 1, 1><<<gHID, 32, 0, stream>>>(t1_bf, W2T, b2, HID, gin_bf, HID, h1f[0], HID);
  k_fanout<<<(BATCH * HID) / 256, 256, 0, stream>>>(h1f[0], gin_bf, h2f[0], h1bf[0], h2bf[0]);

  // ---- 128 serial steps, 6 stream-ordered kernels each
  int cur = 0;
  for (int t = 0; t < T; ++t) {
    const int nxt = cur ^ 1;
    // a = gin @ Wvo + bvo   (folded attention)
    k_gemm<0, 1, 0><<<gHID, 32, 0, stream>>>(gin_bf, WvoT, bvo, HID, a_bf, HID, nullptr, 0);
    // GRU layer 0: h1' = gru(a, h1)
    k_gru<<<gHID, 32, 0, stream>>>(a_bf, h1bf[cur], h1f[cur], Wih0T, Whh0T,
                                   bih0, bhh0, h1f[nxt], h1bf[nxt]);
    // GRU layer 1: h2' = gru(h1', h2)
    k_gru<<<gHID, 32, 0, stream>>>(h1bf[nxt], h2bf[cur], h2f[cur], Wih1T, Whh1T,
                                   bih1, bhh1, h2f[nxt], h2bf[nxt]);
    // nz = h2' @ Wh + bh  -> out[:, t, :] (f32) and nz_bf for the MLP
    k_gemm<0, 1, 1><<<gLAT, 32, 0, stream>>>(h2bf[nxt], WhT, bh, HID,
                                             nz_bf, LAT, outp + (size_t)t * LAT, T * LAT);
    // gin = relu( relu(nz@w1+b1) @ w2 + b2 )
    k_gemm<1, 1, 0><<<gHID, 32, 0, stream>>>(nz_bf, W1T, b1, LAT, t1_bf, HID, nullptr, 0);
    k_gemm<1, 1, 0><<<gHID, 32, 0, stream>>>(t1_bf, W2T, b2, HID, gin_bf, HID, nullptr, 0);
    cur = nxt;
  }
}
